// TemporalAttention_5884105195656
// MI455X (gfx1250) — matrix operand order repaired and checked by reading, compiled, not run on hardware
//
#include <hip/hip_runtime.h>
#include <stdint.h>

typedef _Float16 f16;
typedef _Float16 v16h __attribute__((ext_vector_type(16)));
typedef float    v8f  __attribute__((ext_vector_type(8)));

#define TT     16     // temporal window
#define CC     320    // channels
#define HEADS  8
#define DD     40     // head dim
#define SP     1024   // 32*32 spatial
#define NT_C   20     // CC/16 N-tiles
#define KSTEPS 10     // CC/32 K-steps

__global__ __launch_bounds__(256)
void ta_fused_kernel(const float* __restrict__ x,
                     const float* __restrict__ Wq, const float* __restrict__ Wk,
                     const float* __restrict__ Wv, const float* __restrict__ Wo,
                     const float* __restrict__ bo,
                     const float* __restrict__ relk, const float* __restrict__ relv,
                     float* __restrict__ out)
{
    __shared__ float xin32[TT][CC];          // 20KB input tile f32 (fragments + residual)
    __shared__ f16   qs [TT][CC];            // 10KB
    __shared__ f16   ks [TT][CC];            // 10KB
    __shared__ f16   vs [TT][CC];            // 10KB
    __shared__ f16   abuf[HEADS][TT][32];    // 8KB: first G = Q*RK^T slice, then attn
    __shared__ f16   hout[TT][CC];           // 10KB concatenated head outputs

    const int wg   = blockIdx.x;
    const int bb   = wg >> 10;        // batch
    const int sp   = wg & 1023;       // spatial index xx*32+yy
    const int tid  = threadIdx.x;
    const int lane = tid & 31;
    const int wave = tid >> 5;
    const int hi   = lane >> 4;       // half-wave select
    const int lmod = lane & 15;

    // x index: ((bb*TT+f)*CC + ch)*SP + sp
    const long base = (long)bb * TT * CC * SP + sp;

    // warm L2 for the weights (global_prefetch_b8)
    __builtin_prefetch(Wq, 0, 1);
    __builtin_prefetch(Wk, 0, 1);
    __builtin_prefetch(Wv, 0, 1);
    __builtin_prefetch(Wo, 0, 1);

    // ------- Phase 0: async gather of strided input tile -> LDS (ASYNCcnt path)
    for (int i = tid; i < TT * CC; i += 256) {
        int f = i / CC, ch = i - f * CC;
        unsigned long long gaddr =
            (unsigned long long)(uintptr_t)(x + base + (long)(f * CC + ch) * SP);
        unsigned loff = (unsigned)(uintptr_t)&xin32[f][ch];
        asm volatile("global_load_async_to_lds_b32 %0, %1, off"
                     : : "v"(loff), "v"(gaddr) : "memory");
    }
    asm volatile("s_wait_asynccnt 0" : : : "memory");
    __syncthreads();

    // ------- Phase 1: QKV projections (WMMA) --------------------------------
    v16h afr[KSTEPS];   // A-fragments of the 16x320 input, cached across tiles
    #pragma unroll
    for (int kk = 0; kk < KSTEPS; ++kk) {
        #pragma unroll
        for (int h = 0; h < 16; ++h) {
            int v = h >> 1;
            int K = kk * 32 + hi * 16 + h;
            afr[kk][h] = (f16)xin32[lmod][K];
        }
    }
    for (int tt = wave; tt < 3 * NT_C; tt += 8) {
        const int mat   = tt / NT_C;            // 0=q 1=k 2=v
        const int ncol0 = (tt - mat * NT_C) * 16;
        const float* W  = (mat == 0) ? Wq : (mat == 1) ? Wk : Wv;
        v8f acc = {};
        #pragma unroll
        for (int kk = 0; kk < KSTEPS; ++kk) {
            v16h bfr;
            #pragma unroll
            for (int h = 0; h < 16; ++h) {
                int K = kk * 32 + hi * 16 + h;   // B layout: K striped per half-wave
                bfr[h] = (f16)W[(ncol0 + lmod) * CC + K];
            }
            acc = __builtin_amdgcn_wmma_f32_16x16x32_f16(
                      false, afr[kk], false, bfr, (short)0, acc, false, false);
        }
        #pragma unroll
        for (int r = 0; r < 8; ++r) {
            f16 val = (f16)acc[r];
            int m = r + 8 * hi;
            if (mat == 0)      qs[m][ncol0 + lmod] = val;
            else if (mat == 1) ks[m][ncol0 + lmod] = val;
            else               vs[m][ncol0 + lmod] = val;
        }
    }
    __syncthreads();

    // ------- Phase 2: attention, one wave per head --------------------------
    {
        const int   head  = wave;
        const int   dc    = head * DD;
        const float scale = 0.15811388300841897f;   // 1/sqrt(40)

        // A/B fragments for q and k^T over d (padded 40 -> 64)
        v16h qa[2], kb[2];
        #pragma unroll
        for (int kk = 0; kk < 2; ++kk) {
            #pragma unroll
            for (int h = 0; h < 16; ++h) {
                int v  = h >> 1;
                int Ka = kk * 32 + hi * 16 + h;
                int Kb = kk * 32 + hi * 16 + h;
                qa[kk][h] = (Ka < DD) ? qs[lmod][dc + Ka] : (f16)0.0f;
                kb[kk][h] = (Kb < DD) ? ks[lmod][dc + Kb] : (f16)0.0f;
            }
        }
        v8f sim = {};
        sim = __builtin_amdgcn_wmma_f32_16x16x32_f16(false, qa[0], false, kb[0], (short)0, sim, false, false);
        sim = __builtin_amdgcn_wmma_f32_16x16x32_f16(false, qa[1], false, kb[1], (short)0, sim, false, false);

        // G = Q @ RK^T, shift columns j=0..31 only (j = s-t+16 in [1,31])
        #pragma unroll
        for (int tj = 0; tj < 2; ++tj) {
            v8f g = {};
            #pragma unroll
            for (int kk = 0; kk < 2; ++kk) {
                v16h rb;
                #pragma unroll
                for (int h = 0; h < 16; ++h) {
                    int K = kk * 32 + hi * 16 + h;
                    rb[h] = (K < DD) ? (f16)relk[(tj * 16 + lmod) * DD + K] : (f16)0.0f;
                }
                g = __builtin_amdgcn_wmma_f32_16x16x32_f16(
                        false, qa[kk], false, rb, (short)0, g, false, false);
            }
            #pragma unroll
            for (int r = 0; r < 8; ++r)
                abuf[head][r + 8 * hi][tj * 16 + lmod] = (f16)g[r];
        }

        // sim + gathered rel bias, scale, causal mask (per C-fragment element)
        float e[8];
        #pragma unroll
        for (int r = 0; r < 8; ++r) {
            int t = r + 8 * hi;
            int s = lmod;
            float v = (sim[r] + (float)abuf[head][t][s - t + 16]) * scale;
            e[r] = (s <= t) ? v : -1.0e16f;
        }

        // row-wise softmax over s: reduce across the 16 lanes of each half
        #pragma unroll
        for (int r = 0; r < 8; ++r) {
            float m = e[r];
            #pragma unroll
            for (int off = 8; off >= 1; off >>= 1)
                m = fmaxf(m, __shfl_xor(m, off, 32));
            float p = __expf(e[r] - m);
            float ssum = p;
            #pragma unroll
            for (int off = 8; off >= 1; off >>= 1)
                ssum += __shfl_xor(ssum, off, 32);
            e[r] = p / ssum;
        }

        // overwrite abuf with attn as zero-padded 16x32 f16 A-operand
        #pragma unroll
        for (int r = 0; r < 8; ++r) {
            int t = r + 8 * hi;
            abuf[head][t][lmod]      = (f16)e[r];
            abuf[head][t][lmod + 16] = (f16)0.0f;
        }

        // attn A-fragment (K = s, padded to 32)
        v16h aat;
        // shifted-attn A-fragment A2[t][j] = attn[t][t+j-16]
        v16h a2;
        #pragma unroll
        for (int h = 0; h < 16; ++h) {
            int v  = h >> 1;
            int Ka = hi * 16 + h;  // < 32
            aat[h] = abuf[head][lmod][Ka];
            int s2 = lmod + Ka - 16;
            a2[h] = (s2 >= 0 && s2 < TT) ? abuf[head][lmod][s2] : (f16)0.0f;
        }

        // out = A2 @ RV + attn @ v, 3 N-tiles covering d = 40
        #pragma unroll
        for (int tile = 0; tile < 3; ++tile) {
            const int dd = tile * 16 + lmod;
            v16h vb, rvb;
            #pragma unroll
            for (int h = 0; h < 16; ++h) {
                int K = hi * 16 + h;     // B-layout K index, 0..31
                vb[h]  = (K < TT && dd < DD) ? vs[K][dc + dd] : (f16)0.0f;
                rvb[h] = (dd < DD) ? (f16)relv[K * DD + dd] : (f16)0.0f;
            }
            v8f z = {};
            v8f o = __builtin_amdgcn_wmma_f32_16x16x32_f16(
                        false, a2, false, rvb, (short)0, z, false, false);
            o = __builtin_amdgcn_wmma_f32_16x16x32_f16(
                        false, aat, false, vb, (short)0, o, false, false);
            if (dd < DD) {
                #pragma unroll
                for (int r = 0; r < 8; ++r)
                    hout[r + 8 * hi][dc + dd] = (f16)o[r];
            }
        }
    }
    __syncthreads();

    // ------- Phase 3: output projection + bias + residual (from LDS) -------
    v16h hfr[KSTEPS];
    #pragma unroll
    for (int kk = 0; kk < KSTEPS; ++kk) {
        #pragma unroll
        for (int h = 0; h < 16; ++h) {
            int v = h >> 1;
            int K = kk * 32 + hi * 16 + h;
            hfr[kk][h] = hout[lmod][K];
        }
    }
    for (int tile = wave; tile < NT_C; tile += 8) {
        const int ncol0 = tile * 16;
        v8f acc = {};
        #pragma unroll
        for (int kk = 0; kk < KSTEPS; ++kk) {
            v16h bfr;
            #pragma unroll
            for (int h = 0; h < 16; ++h) {
                int K = kk * 32 + hi * 16 + h;
                bfr[h] = (f16)Wo[(ncol0 + lmod) * CC + K];
            }
            acc = __builtin_amdgcn_wmma_f32_16x16x32_f16(
                      false, hfr[kk], false, bfr, (short)0, acc, false, false);
        }
        const int   col  = ncol0 + lmod;
        const float bias = bo[col];
        #pragma unroll
        for (int r = 0; r < 8; ++r) {
            int  f  = r + 8 * hi;
            long gi = base + (long)(f * CC + col) * SP;
            out[gi] = acc[r] + bias + xin32[f][col];   // residual from LDS, not HBM
        }
    }
}

extern "C" void kernel_launch(void* const* d_in, const int* in_sizes, int n_in,
                              void* d_out, int out_size, void* d_ws, size_t ws_size,
                              hipStream_t stream) {
    const float* x    = (const float*)d_in[0];
    const float* Wq   = (const float*)d_in[1];
    const float* Wk   = (const float*)d_in[2];
    const float* Wv   = (const float*)d_in[3];
    const float* Wo   = (const float*)d_in[4];
    const float* bo   = (const float*)d_in[5];
    const float* relk = (const float*)d_in[6];
    const float* relv = (const float*)d_in[7];
    float* o = (float*)d_out;

    // one workgroup per (batch, spatial) temporal window: 4 * 1024 = 4096
    ta_fused_kernel<<<dim3(4096), dim3(256), 0, stream>>>(
        x, Wq, Wk, Wv, Wo, bo, relk, relv, o);
    (void)d_ws; (void)ws_size; (void)in_sizes; (void)n_in; (void)out_size;
}
